// GraphConvolution_79809082294822
// MI455X (gfx1250) — matrix operand order repaired and checked by reading, not compiled, not run on hardware
//
#include <hip/hip_runtime.h>
#include <math.h>

#define D_IN  256
#define D_OUT 128
#define LDA   264   // padded LDS row stride (shorts) to spread banks

typedef __attribute__((ext_vector_type(16))) __bf16 v16bf;
typedef __attribute__((ext_vector_type(8)))  float  v8f;

union ABfrag { unsigned int u[8]; v16bf v; };

__device__ __forceinline__ unsigned short f2bf(float f) {
  union { float f; unsigned int u; } v; v.f = f;
  unsigned int u = v.u;
  u += 0x7FFFu + ((u >> 16) & 1u);   // round-to-nearest-even
  return (unsigned short)(u >> 16);
}

// Transpose W [256x128] f32 -> Wt [128x256] bf16 (K contiguous per output column)
__global__ __launch_bounds__(256) void wconvert_kernel(const float* __restrict__ W,
                                                       unsigned short* __restrict__ Wt) {
  int i = blockIdx.x * 256 + threadIdx.x;
  if (i < D_IN * D_OUT) {
    int k = i / D_OUT, c = i % D_OUT;
    Wt[c * D_IN + k] = f2bf(W[i]);
  }
}

// Out[16 rows x 128 cols per block] = X @ W  via v_wmma_f32_16x16x32_bf16.
// 256 threads = 8 wave32; wave w owns column tile w. K loop: 256/32 = 8 WMMAs.
__global__ __launch_bounds__(256) void gemm_bf16_kernel(const float* __restrict__ X,
                                                        const unsigned short* __restrict__ Wt,
                                                        float* __restrict__ Out,
                                                        int nrows, int doRelu) {
  __shared__ unsigned short ldsA[16 * LDA];
  const int tid = threadIdx.x;
  const int rowBase = blockIdx.x * 16;

  // Stage 16x256 X tile into LDS as bf16 (coalesced f32 reads)
  for (int i = tid; i < 16 * D_IN; i += 256) {
    int r = i >> 8;
    int k = i & (D_IN - 1);
    int gr = rowBase + r;
    if (gr >= nrows) gr = nrows - 1;           // clamp (harmless duplicate)
    ldsA[r * LDA + k] = f2bf(X[(size_t)gr * D_IN + k]);
  }
  __syncthreads();

  const int wave = tid >> 5;
  const int lane = tid & 31;
  const int half = lane >> 4;                  // 0: lanes 0-15, 1: lanes 16-31
  const int l16  = lane & 15;
  const int colBase = wave * 16;

  v8f acc = {};
  const unsigned short* wcol = Wt + (size_t)(colBase + l16) * D_IN;

  for (int k0 = 0; k0 < D_IN; k0 += 32) {
    ABfrag a, b;
#pragma unroll
    for (int v = 0; v < 8; ++v) {
      // A 16x32 bf16 layout: lanes0-15 K {0..7,16..23}, lanes16-31 K {8..15,24..31}
      int ka = ((v < 4) ? (2 * v) : (16 + 2 * (v - 4))) + (half ? 8 : 0);
      a.u[v] = *(const unsigned int*)&ldsA[l16 * LDA + k0 + ka];
      // B 32x16 bf16 layout: lanes0-15 K {0..7,16..23}, lanes16-31 K {8..15,24..31}, lane group = N
      int kb = ((v < 4) ? (2 * v) : (16 + 2 * (v - 4))) + (half ? 8 : 0);
      b.u[v] = *(const unsigned int*)&wcol[k0 + kb];
    }
    acc = __builtin_amdgcn_wmma_f32_16x16x32_bf16(false, a.v, false, b.v,
                                                  (short)0, acc, false, false);
  }

#pragma unroll
  for (int r = 0; r < 8; ++r) {
    int row = rowBase + r + half * 8;          // C/D: VGPR r -> M=r (+8 upper half)
    if (row < nrows) {
      float val = acc[r];
      if (doRelu) val = fmaxf(val, 0.0f);
      Out[(size_t)row * D_OUT + colBase + l16] = val;
    }
  }
}

// One wave32 per edge: gather 128 feats of src row, weight, atomic-add into dst row.
__global__ __launch_bounds__(256) void spmm_kernel(const int* __restrict__ src,
                                                   const int* __restrict__ dst,
                                                   const float* __restrict__ ew,
                                                   const float* __restrict__ H,
                                                   float* __restrict__ Acc,
                                                   int nEdges) {
  const int wave = threadIdx.x >> 5;
  const int lane = threadIdx.x & 31;
  const int e = blockIdx.x * 8 + wave;
  if (e >= nEdges) return;
  const int   s = src[e];
  const int   d = dst[e];
  const float w = ew[e];
  const float4 hv = *(const float4*)(H + (size_t)s * D_OUT + lane * 4);
  float* ap = Acc + (size_t)d * D_OUT + lane * 4;
  atomicAdd(ap + 0, w * hv.x);
  atomicAdd(ap + 1, w * hv.y);
  atomicAdd(ap + 2, w * hv.z);
  atomicAdd(ap + 3, w * hv.w);
}

__device__ __forceinline__ float waveSum(float v) {
#pragma unroll
  for (int off = 16; off > 0; off >>= 1) v += __shfl_xor(v, off, 32);
  return v;
}

// One wave32 per node: relu, 3 dot-scores, sigmoid, 3x3 mix, softmax(T=3), blend.
__global__ __launch_bounds__(256) void finalize_kernel(const float* __restrict__ accL,
                                                       const float* __restrict__ accH,
                                                       const float* __restrict__ hM,
                                                       const float* __restrict__ aL,
                                                       const float* __restrict__ aH,
                                                       const float* __restrict__ aM,
                                                       const float* __restrict__ attv,
                                                       float* __restrict__ out,
                                                       int nNodes) {
  const int wave = threadIdx.x >> 5;
  const int lane = threadIdx.x & 31;
  const int node = blockIdx.x * 8 + wave;
  if (node >= nNodes) return;
  const size_t base = (size_t)node * D_OUT + lane * 4;

  float4 hl = *(const float4*)(accL + base);
  float4 hh = *(const float4*)(accH + base);
  float4 hm = *(const float4*)(hM + base);
  hl.x = fmaxf(hl.x, 0.f); hl.y = fmaxf(hl.y, 0.f); hl.z = fmaxf(hl.z, 0.f); hl.w = fmaxf(hl.w, 0.f);
  hh.x = fmaxf(hh.x, 0.f); hh.y = fmaxf(hh.y, 0.f); hh.z = fmaxf(hh.z, 0.f); hh.w = fmaxf(hh.w, 0.f);

  const float4 al = *(const float4*)(aL + lane * 4);
  const float4 ah = *(const float4*)(aH + lane * 4);
  const float4 am = *(const float4*)(aM + lane * 4);

  float s0 = waveSum(hl.x*al.x + hl.y*al.y + hl.z*al.z + hl.w*al.w);
  float s1 = waveSum(hh.x*ah.x + hh.y*ah.y + hh.z*ah.z + hh.w*ah.w);
  float s2 = waveSum(hm.x*am.x + hm.y*am.y + hm.z*am.z + hm.w*am.w);

  float g0 = 1.0f / (1.0f + __expf(-s0));
  float g1 = 1.0f / (1.0f + __expf(-s1));
  float g2 = 1.0f / (1.0f + __expf(-s2));

  const float inv3 = 1.0f / 3.0f;
  float z0 = (g0*attv[0] + g1*attv[3] + g2*attv[6]) * inv3;
  float z1 = (g0*attv[1] + g1*attv[4] + g2*attv[7]) * inv3;
  float z2 = (g0*attv[2] + g1*attv[5] + g2*attv[8]) * inv3;
  float m  = fmaxf(z0, fmaxf(z1, z2));
  float e0 = __expf(z0 - m), e1 = __expf(z1 - m), e2 = __expf(z2 - m);
  float sc = 3.0f / (e0 + e1 + e2);            // folds the final *3.0
  float a0 = e0 * sc, a1 = e1 * sc, a2 = e2 * sc;

  float4 o;
  o.x = a0*hl.x + a1*hh.x + a2*hm.x;
  o.y = a0*hl.y + a1*hh.y + a2*hm.y;
  o.z = a0*hl.z + a1*hh.z + a2*hm.z;
  o.w = a0*hl.w + a1*hh.w + a2*hm.w;
  *(float4*)(out + base) = o;
}

extern "C" void kernel_launch(void* const* d_in, const int* in_sizes, int n_in,
                              void* d_out, int out_size, void* d_ws, size_t ws_size,
                              hipStream_t stream) {
  const float* x     = (const float*)d_in[0];
  const int*   esrc  = (const int*)d_in[1];
  const int*   edst  = (const int*)d_in[2];
  const float* wlow  = (const float*)d_in[3];
  const float* whigh = (const float*)d_in[4];
  // d_in[5] = w_unnorm (unused by reference path)
  const float* Wl    = (const float*)d_in[6];
  const float* Wh    = (const float*)d_in[7];
  const float* Wm    = (const float*)d_in[8];
  const float* aL    = (const float*)d_in[9];
  const float* aH    = (const float*)d_in[10];
  const float* aM    = (const float*)d_in[11];
  const float* attv  = (const float*)d_in[12];

  const int N = in_sizes[0] / D_IN;   // 100000
  const int E = in_sizes[1];          // 1600000
  float* out = (float*)d_out;

  char* ws = (char*)d_ws;
  const size_t wbytes = (size_t)D_IN * D_OUT * sizeof(unsigned short);
  unsigned short* WtL = (unsigned short*)(ws + 0 * wbytes);
  unsigned short* WtH = (unsigned short*)(ws + 1 * wbytes);
  unsigned short* WtM = (unsigned short*)(ws + 2 * wbytes);
  const size_t mb = (size_t)N * D_OUT * sizeof(float);
  float* tmpL = (float*)(ws + 3 * wbytes);
  float* tmpH = (float*)((char*)tmpL + mb);
  float* hM   = (float*)((char*)tmpH + mb);
  float* accH = (float*)((char*)hM + mb);
  float* accL = out;   // reuse output buffer as low-branch accumulator

  // zero accumulators every call (harness poisons buffers)
  hipMemsetAsync(accL, 0, mb, stream);
  hipMemsetAsync(accH, 0, mb, stream);

  const int cblocks = (D_IN * D_OUT + 255) / 256;
  wconvert_kernel<<<cblocks, 256, 0, stream>>>(Wl, WtL);
  wconvert_kernel<<<cblocks, 256, 0, stream>>>(Wh, WtH);
  wconvert_kernel<<<cblocks, 256, 0, stream>>>(Wm, WtM);

  const int gblocks = (N + 15) / 16;
  gemm_bf16_kernel<<<gblocks, 256, 0, stream>>>(x, WtL, tmpL, N, 0);
  gemm_bf16_kernel<<<gblocks, 256, 0, stream>>>(x, WtH, tmpH, N, 0);
  gemm_bf16_kernel<<<gblocks, 256, 0, stream>>>(x, WtM, hM,   N, 1);

  const int eblocks = (E + 7) / 8;
  spmm_kernel<<<eblocks, 256, 0, stream>>>(esrc, edst, wlow,  tmpL, accL, E);
  spmm_kernel<<<eblocks, 256, 0, stream>>>(esrc, edst, whigh, tmpH, accH, E);

  const int nblocks = (N + 7) / 8;
  finalize_kernel<<<nblocks, 256, 0, stream>>>(accL, accH, hM, aL, aH, aM, attv, out, N);
}
